// SelfAttentionBlock_57071525429440
// MI455X (gfx1250) — matrix operand order repaired and checked by reading, not compiled, not run on hardware
//
#include <hip/hip_runtime.h>
#include <hip/hip_bf16.h>

// ---------------------------------------------------------------------------
// Self-attention (B=8, T=2048, C=1024), bf16 WMMA pipeline for gfx1250.
//   1) convert x, Wq, Wk, Wv to bf16
//   2) Q = x Wq^T, K = x Wk^T (bf16), V^T = (x Wv^T)^T (bf16)
//   3) S = Q K^T / 32, causal-masked (bf16)
//   4) row softmax in-place on S
//   5) out = P V (K-contiguous vs V^T), fp32
// GEMM: 128x128 block tile, 8 waves of 32x64, BK=64 (2 WMMA K-steps/iter),
// double-buffered LDS filled with GLOBAL_LOAD_ASYNC_TO_LDS_B128 (ASYNCcnt).
// ---------------------------------------------------------------------------

typedef __bf16        bfx16 __attribute__((ext_vector_type(16)));
typedef float         fx8   __attribute__((ext_vector_type(8)));
typedef unsigned int  ux4   __attribute__((ext_vector_type(4)));

#define BMT  128   // block M tile
#define BNT  128   // block N tile
#define BKT  64    // K per iteration (two 16x16x32 WMMA K-steps)
#define LDT  72    // padded bf16 elems per LDS tile row (144B, 16B-aligned, conflict-free)
#define TILE_ELEMS (128 * LDT)   // 9216 elements = 18432 B per tile buffer

__device__ __forceinline__ unsigned short f2bf(float f) {
    unsigned int u = __float_as_uint(f);
    u += 0x7FFFu + ((u >> 16) & 1u);   // round-to-nearest-even
    return (unsigned short)(u >> 16);
}
__device__ __forceinline__ float bf2f(unsigned short h) {
    return __uint_as_float(((unsigned int)h) << 16);
}

// Issue one 16B async global->LDS copy. LDS byte offset = low 32 bits of the
// generic (flat) address of the __shared__ destination (ISA: LDS aperture maps
// addr[31:0] to the LDS offset). Tracked by ASYNCcnt.
__device__ __forceinline__ void async_b128(const unsigned short* lds_dst, const void* gsrc) {
    unsigned loff = (unsigned)(unsigned long long)lds_dst;
    asm volatile("global_load_async_to_lds_b128 %0, %1, off"
                 :: "v"(loff), "v"((unsigned long long)gsrc)
                 : "memory");
}
__device__ __forceinline__ void wait_async0() {
    asm volatile("s_wait_asynccnt 0x0" ::: "memory");
}

// ---------------------------------------------------------------------------
// fp32 -> bf16 conversion, 4 elements / thread
// ---------------------------------------------------------------------------
__global__ __launch_bounds__(256)
void f32_to_bf16_kernel(const float* __restrict__ in, unsigned short* __restrict__ out,
                        long long n) {
    long long i = ((long long)blockIdx.x * 256 + threadIdx.x) * 4;
    if (i >= n) return;                  // sizes are exact multiples of 4
    float4 f = *(const float4*)(in + i);
    unsigned long long p = (unsigned long long)f2bf(f.x)
                         | ((unsigned long long)f2bf(f.y) << 16)
                         | ((unsigned long long)f2bf(f.z) << 32)
                         | ((unsigned long long)f2bf(f.w) << 48);
    *(unsigned long long*)(out + i) = p;
}

// ---------------------------------------------------------------------------
// Tiled bf16 GEMM:  C[m,n] = alpha * sum_k A[m,k] * B[n,k]
//   A: M x K row-major bf16 (K contiguous), B: N x K row-major bf16
// MODE 0: store bf16            C[b] + m*N + n
// MODE 1: store bf16 transposed-by-batch: Vt[(m/Tb)*N + n][t=m%Tb]  (V^T layout)
// MODE 2: scores: val*alpha, causal mask (n>m -> -inf), bf16; masked tiles skip K loop
// MODE 3: out: fp32 store, K loop truncated to keys <= row-tile end (causal P.V)
// ---------------------------------------------------------------------------
template <int MODE>
__global__ __launch_bounds__(256)
void gemm_bf16_kernel(const unsigned short* __restrict__ A,
                      const unsigned short* __restrict__ B,
                      void* __restrict__ Cv,
                      int N, int Kdim,
                      long long sA, long long sB, long long sC,
                      float alpha, int Tb) {
    const int tid   = threadIdx.x;
    const int lane  = tid & 31;
    const int wid   = tid >> 5;            // 8 waves (wave32)
    const int tileM = blockIdx.y * BMT;
    const int tileN = blockIdx.x * BNT;
    const int b     = blockIdx.z;

    A += (size_t)b * (size_t)sA;
    B += (size_t)b * (size_t)sB;

    __shared__ __align__(16) unsigned short As[2 * TILE_ELEMS];
    __shared__ __align__(16) unsigned short Bs[2 * TILE_ELEMS];

    fx8 acc[2][4];
    #pragma unroll
    for (int i = 0; i < 2; ++i)
        #pragma unroll
        for (int j = 0; j < 4; ++j) acc[i][j] = (fx8){};

    int nkt = Kdim / BKT;
    if (MODE == 2 && tileN > tileM + (BMT - 1)) nkt = 0;          // fully masked tile
    if (MODE == 3) { int lim = (tileM + BMT) / BKT; if (lim < nkt) nkt = lim; }

    // async-fill assignment: 128 rows x 64 cols = 1024 16B chunks, 4 per thread
    const int lrow = tid >> 3;             // rows 0..31 (+32 per chunk step)
    const int lcol = (tid & 7) * 8;        // bf16 column offset within BK

    const int mw = (wid & 3) * 32;         // wave subtile: 32 rows x 64 cols
    const int nw = (wid >> 2) * 64;

    auto issue_tile = [&](const unsigned short* __restrict__ G, int rowBase,
                          unsigned short* ldsBuf, int k0) {
        #pragma unroll
        for (int c = 0; c < 4; ++c) {
            const int row = lrow + c * 32;
            async_b128(ldsBuf + row * LDT + lcol,
                       G + (size_t)(rowBase + row) * Kdim + k0 + lcol);
        }
    };

    // LDS fragment loaders in contiguous k order (K = 16*half + s), same for A and B
    auto ldA = [&](const unsigned short* Ab, int mbase, int kk) -> bfx16 {
        const unsigned short* p = Ab + (size_t)(mbase + (lane & 15)) * LDT
                                     + kk * 32 + (lane >> 4) * 16;
        union { bfx16 v; ux4 u[2]; } r;
        r.u[0] = *(const ux4*)p;           // K = 16*half .. +7
        r.u[1] = *(const ux4*)(p + 8);     // K = 16*half+8 .. +15
        return r.v;
    };
    auto ldB = [&](const unsigned short* Bb, int nbase, int kk) -> bfx16 {
        const unsigned short* p = Bb + (size_t)(nbase + (lane & 15)) * LDT
                                     + kk * 32 + (lane >> 4) * 16;
        union { bfx16 v; ux4 u[2]; } r;
        r.u[0] = *(const ux4*)p;           // K = 16*half .. +7
        r.u[1] = *(const ux4*)(p + 8);     // K = 16*half+8 .. +15
        return r.v;
    };

    if (nkt > 0) {                         // prologue: tile 0 -> buffer 0
        issue_tile(A, tileM, As, 0);
        issue_tile(B, tileN, Bs, 0);
    }

    for (int kt = 0; kt < nkt; ++kt) {
        wait_async0();                     // our tile-kt chunks have landed
        __syncthreads();                   // everyone's landed; next buf free
        const int cur = kt & 1;
        if (kt + 1 < nkt) {                // overlap: fill other buffer
            const int k0 = (kt + 1) * BKT;
            issue_tile(A, tileM, As + (cur ^ 1) * TILE_ELEMS, k0);
            issue_tile(B, tileN, Bs + (cur ^ 1) * TILE_ELEMS, k0);
        }
        const unsigned short* Ab = As + cur * TILE_ELEMS;
        const unsigned short* Bb = Bs + cur * TILE_ELEMS;
        #pragma unroll
        for (int kk = 0; kk < 2; ++kk) {
            bfx16 a0 = ldA(Ab, mw + 0,  kk);
            bfx16 a1 = ldA(Ab, mw + 16, kk);
            bfx16 w0 = ldB(Bb, nw + 0,  kk);
            bfx16 w1 = ldB(Bb, nw + 16, kk);
            bfx16 w2 = ldB(Bb, nw + 32, kk);
            bfx16 w3 = ldB(Bb, nw + 48, kk);
            acc[0][0] = __builtin_amdgcn_wmma_f32_16x16x32_bf16(false, a0, false, w0, (short)0, acc[0][0], false, false);
            acc[0][1] = __builtin_amdgcn_wmma_f32_16x16x32_bf16(false, a0, false, w1, (short)0, acc[0][1], false, false);
            acc[0][2] = __builtin_amdgcn_wmma_f32_16x16x32_bf16(false, a0, false, w2, (short)0, acc[0][2], false, false);
            acc[0][3] = __builtin_amdgcn_wmma_f32_16x16x32_bf16(false, a0, false, w3, (short)0, acc[0][3], false, false);
            acc[1][0] = __builtin_amdgcn_wmma_f32_16x16x32_bf16(false, a1, false, w0, (short)0, acc[1][0], false, false);
            acc[1][1] = __builtin_amdgcn_wmma_f32_16x16x32_bf16(false, a1, false, w1, (short)0, acc[1][1], false, false);
            acc[1][2] = __builtin_amdgcn_wmma_f32_16x16x32_bf16(false, a1, false, w2, (short)0, acc[1][2], false, false);
            acc[1][3] = __builtin_amdgcn_wmma_f32_16x16x32_bf16(false, a1, false, w3, (short)0, acc[1][3], false, false);
        }
    }

    // epilogue: C/D layout -> VGPR v: lanes 0-15 M=v, lanes 16-31 M=v+8; N=lane%16
    const int mofs = (lane >> 4) * 8;
    const int ncol = lane & 15;
    #pragma unroll
    for (int ms = 0; ms < 2; ++ms) {
        #pragma unroll
        for (int ns = 0; ns < 4; ++ns) {
            fx8 a = acc[ms][ns];
            #pragma unroll
            for (int v = 0; v < 8; ++v) {
                const int gm = tileM + mw + ms * 16 + mofs + v;
                const int gn = tileN + nw + ns * 16 + ncol;
                float val = a[v] * alpha;
                if (MODE == 3) {
                    float* Cf = (float*)Cv;
                    Cf[(size_t)b * (size_t)sC + (size_t)gm * N + gn] = val;
                } else {
                    unsigned short* Ch = (unsigned short*)Cv;
                    unsigned short h;
                    if (MODE == 2 && gn > gm) h = 0xFF80u;     // bf16 -inf
                    else                      h = f2bf(val);
                    size_t idx;
                    if (MODE == 1) idx = ((size_t)(gm / Tb) * N + gn) * (size_t)Tb + (size_t)(gm % Tb);
                    else           idx = (size_t)b * (size_t)sC + (size_t)gm * N + gn;
                    Ch[idx] = h;
                }
            }
        }
    }
}

// ---------------------------------------------------------------------------
// In-place row softmax over bf16 scores (row length T = 2048, 8 elems/thread)
// ---------------------------------------------------------------------------
__global__ __launch_bounds__(256)
void softmax_rows_kernel(unsigned short* __restrict__ S, int T) {
    const int tid  = threadIdx.x;
    const int lane = tid & 31;
    const int wid  = tid >> 5;
    unsigned short* rp = S + (size_t)blockIdx.x * T;

    union { ux4 u; unsigned short h[8]; } d;
    d.u = *(const ux4*)(rp + tid * 8);

    float v[8];
    float m = -3.0e38f;
    #pragma unroll
    for (int i = 0; i < 8; ++i) { v[i] = bf2f(d.h[i]); m = fmaxf(m, v[i]); }

    #pragma unroll
    for (int off = 16; off > 0; off >>= 1) m = fmaxf(m, __shfl_xor(m, off, 32));

    __shared__ float red[8];
    if (lane == 0) red[wid] = m;
    __syncthreads();
    m = red[0];
    #pragma unroll
    for (int i = 1; i < 8; ++i) m = fmaxf(m, red[i]);
    __syncthreads();

    float s = 0.0f;
    #pragma unroll
    for (int i = 0; i < 8; ++i) { v[i] = __expf(v[i] - m); s += v[i]; }
    #pragma unroll
    for (int off = 16; off > 0; off >>= 1) s += __shfl_xor(s, off, 32);
    if (lane == 0) red[wid] = s;
    __syncthreads();
    s = red[0] + red[1] + red[2] + red[3] + red[4] + red[5] + red[6] + red[7];

    const float inv = 1.0f / s;
    #pragma unroll
    for (int i = 0; i < 8; ++i) d.h[i] = f2bf(v[i] * inv);
    *(ux4*)(rp + tid * 8) = d.u;
}

// ---------------------------------------------------------------------------
extern "C" void kernel_launch(void* const* d_in, const int* in_sizes, int n_in,
                              void* d_out, int out_size, void* d_ws, size_t ws_size,
                              hipStream_t stream) {
    (void)in_sizes; (void)n_in; (void)out_size; (void)ws_size;

    const float* x  = (const float*)d_in[0];
    const float* Wq = (const float*)d_in[1];
    const float* Wk = (const float*)d_in[2];
    const float* Wv = (const float*)d_in[3];
    float* out = (float*)d_out;

    const int       Bb = 8, T = 2048, C = 1024;
    const long long MT = (long long)Bb * T;     // 16384 rows total

    // workspace carving (bf16 buffers), ~208 MB total
    char* w = (char*)d_ws;
    unsigned short* xb  = (unsigned short*)w; w += MT * C * 2;
    unsigned short* wqb = (unsigned short*)w; w += (long long)C * C * 2;
    unsigned short* wkb = (unsigned short*)w; w += (long long)C * C * 2;
    unsigned short* wvb = (unsigned short*)w; w += (long long)C * C * 2;
    unsigned short* Qb  = (unsigned short*)w; w += MT * C * 2;
    unsigned short* Kb  = (unsigned short*)w; w += MT * C * 2;
    unsigned short* Vtb = (unsigned short*)w; w += MT * C * 2;      // V^T: [B][C][T]
    unsigned short* S   = (unsigned short*)w; w += (long long)Bb * T * T * 2;

    // 1) convert to bf16
    f32_to_bf16_kernel<<<(int)(MT * C / 1024), 256, 0, stream>>>(x,  xb,  MT * C);
    f32_to_bf16_kernel<<<(int)((long long)C * C / 1024), 256, 0, stream>>>(Wq, wqb, (long long)C * C);
    f32_to_bf16_kernel<<<(int)((long long)C * C / 1024), 256, 0, stream>>>(Wk, wkb, (long long)C * C);
    f32_to_bf16_kernel<<<(int)((long long)C * C / 1024), 256, 0, stream>>>(Wv, wvb, (long long)C * C);

    dim3 blk(256);

    // 2) QKV projections: M=16384, N=C, K=C
    dim3 gq(C / BNT, (int)(MT / BMT), 1);
    gemm_bf16_kernel<0><<<gq, blk, 0, stream>>>(xb, wqb, Qb,  C, C, 0, 0, 0, 1.0f, T);
    gemm_bf16_kernel<0><<<gq, blk, 0, stream>>>(xb, wkb, Kb,  C, C, 0, 0, 0, 1.0f, T);
    gemm_bf16_kernel<1><<<gq, blk, 0, stream>>>(xb, wvb, Vtb, C, C, 0, 0, 0, 1.0f, T);

    // 3) scores: per batch M=T, N=T, K=C, alpha = 1/sqrt(C) = 1/32, causal mask
    dim3 gs(T / BNT, T / BMT, Bb);
    gemm_bf16_kernel<2><<<gs, blk, 0, stream>>>(Qb, Kb, S, T, C,
                                                (long long)T * C, (long long)T * C,
                                                (long long)T * T, 0.03125f, T);

    // 4) softmax rows (in-place on S)
    softmax_rows_kernel<<<(int)MT, 256, 0, stream>>>(S, T);

    // 5) out = P @ V: per batch M=T, N=C, K=T (keys), fp32 output
    dim3 go(C / BNT, T / BMT, Bb);
    gemm_bf16_kernel<3><<<go, blk, 0, stream>>>(S, Vtb, out, C, T,
                                                (long long)T * T, (long long)C * T,
                                                (long long)T * C, 1.0f, T);
}
